// VGG16RoIHead_15255723835520
// MI455X (gfx1250) — matrix operand order repaired and checked by reading, not compiled, not run on hardware
//
#include <hip/hip_runtime.h>
#include <hip/hip_bf16.h>

// ---------------------------------------------------------------------------
// MI455X (gfx1250) VGG16 RoI head.
// Bandwidth-bound on W1/W2 (475 MB fp32 -> ~20us @ 23.3 TB/s). Strategy:
// stream fp32 weights once, convert to bf16 in-register, and push everything
// through v_wmma_f32_16x16x32_bf16 (wave32, 16x16 f32 acc = 8 VGPRs).
// Activations (6.4 MB bf16 feat) stay resident in the 192 MB L2 across
// N-tiles, so HBM traffic stays at the 1x-weights floor.
// ---------------------------------------------------------------------------

typedef __attribute__((ext_vector_type(16))) __bf16 v16bf;
typedef __attribute__((ext_vector_type(8)))  float  v8f;

#define RR   128          // ROIs (M of the GEMMs)
#define CC   512
#define HH   38
#define WW   38
#define PP   7
#define D1   25088        // 512*7*7
#define D2   4096
#define NLOC 84           // 21*4
#define NSC  21

// ---------------------------------------------------------------------------
// RoI max pool (torchvision RoIPool semantics), emits feat directly as bf16.
// One thread per (roi, channel, ph, pw). x is only 5.9 MB -> cache-resident.
// ---------------------------------------------------------------------------
__global__ __launch_bounds__(256) void roi_pool_kernel(
    const float* __restrict__ x, const float* __restrict__ rois,
    const int* __restrict__ roi_idx, __bf16* __restrict__ featB)
{
    int t = blockIdx.x * blockDim.x + threadIdx.x;
    const int total = RR * CC * PP * PP;
    if (t >= total) return;

    int r   = t / (CC * PP * PP);
    int rem = t - r * (CC * PP * PP);
    int c   = rem / (PP * PP);
    int p   = rem - c * (PP * PP);
    int ph  = p / PP;
    int pw  = p - ph * PP;

    // boxes = rois[:, [1,0,3,2]] -> (xmin, ymin, xmax, ymax)
    float xmin = rois[r * 4 + 1], ymin = rois[r * 4 + 0];
    float xmax = rois[r * 4 + 3], ymax = rois[r * 4 + 2];
    // jnp.round == round-half-even -> rintf under default RNE mode
    int x1 = (int)rintf(xmin * 0.0625f);
    int y1 = (int)rintf(ymin * 0.0625f);
    int x2 = (int)rintf(xmax * 0.0625f);
    int y2 = (int)rintf(ymax * 0.0625f);

    float roi_w = (float)max(x2 - x1 + 1, 1);
    float roi_h = (float)max(y2 - y1 + 1, 1);

    int hs = min(max((int)floorf((float)ph        * (roi_h / (float)PP)) + y1, 0), HH);
    int he = min(max((int)ceilf (((float)ph + 1.f) * (roi_h / (float)PP)) + y1, 0), HH);
    int ws = min(max((int)floorf((float)pw        * (roi_w / (float)PP)) + x1, 0), WW);
    int we = min(max((int)ceilf (((float)pw + 1.f) * (roi_w / (float)PP)) + x1, 0), WW);

    float v = 0.0f;
    if (he > hs && we > ws) {
        int b = roi_idx[r];
        const float* xp = x + ((size_t)b * CC + c) * (HH * WW);
        float m = -INFINITY;
        for (int h = hs; h < he; ++h) {
            const float* row = xp + h * WW;
            for (int w = ws; w < we; ++w) m = fmaxf(m, row[w]);
        }
        v = m;
    }
    featB[(size_t)r * D1 + (size_t)c * (PP * PP) + p] = (__bf16)v;
}

// ---------------------------------------------------------------------------
// WMMA GEMM: out[M=128, N] = relu(A[128,K](bf16) * W[N,K](fp32->bf16)^T + b)
// Block = 256 threads = 8 waves; wave w owns M-tile [16w, 16w+16);
// block owns N-tile [16*blockIdx.x, +16). K loop in steps of 32.
// Fragment layouts per CDNA5 ISA 7.12.2 (wave32):
//   A 16x32 bf16: lane l<16 -> row l, K {0..7} in v0-3, {16..23} in v4-7;
//                 lane l+16 -> row l, K {8..15}, {24..31}.
//   B 32x16 bf16: lane owns column n = lane&15; same K split as A:
//                 lanes 0-15 K {0..7},{16..23}; lanes 16-31 K {8..15},{24..31}.
//   C/D 16x16 f32: acc[i] = (row m0 + 8*half + i, col n0 + (lane&15)).
// ---------------------------------------------------------------------------
__global__ __launch_bounds__(256) void gemm_ws(
    const __bf16* __restrict__ A, const float* __restrict__ W,
    const float* __restrict__ bias, __bf16* __restrict__ outB,
    float* __restrict__ outF, int K, int N, int relu)
{
    const int lane = threadIdx.x & 31;
    const int wave = threadIdx.x >> 5;
    const int l    = lane & 15;
    const int half = lane >> 4;

    const int m0 = wave * 16;
    const int n0 = blockIdx.x * 16;
    const int n  = n0 + l;

    const __bf16* arow = A + (size_t)(m0 + l) * K;   // row of A for this lane
    const float*  wrow = W + (size_t)n * K;          // weight row n (B column)

    v8f acc = {};

    for (int k0 = 0; k0 < K; k0 += 32) {
        // ---- A fragment: two 16-byte contiguous chunks of bf16 ----
        const uint4 a_lo = *(const uint4*)(arow + k0 + half * 8);
        const uint4 a_hi = *(const uint4*)(arow + k0 + 16 + half * 8);
        v16bf afrag;
        ((uint4*)&afrag)[0] = a_lo;
        ((uint4*)&afrag)[1] = a_hi;

        // ---- B fragment: two 8-float contiguous chunks of fp32 -> bf16 ----
        const float4* wp = (const float4*)(wrow + k0 + half * 8);
        const float4* wq = (const float4*)(wrow + k0 + 16 + half * 8);
        float4 w0 = wp[0], w1 = wp[1], w2 = wq[0], w3 = wq[1];
        v16bf bfrag;
        bfrag[0]  = (__bf16)w0.x; bfrag[1]  = (__bf16)w0.y;
        bfrag[2]  = (__bf16)w0.z; bfrag[3]  = (__bf16)w0.w;
        bfrag[4]  = (__bf16)w1.x; bfrag[5]  = (__bf16)w1.y;
        bfrag[6]  = (__bf16)w1.z; bfrag[7]  = (__bf16)w1.w;
        bfrag[8]  = (__bf16)w2.x; bfrag[9]  = (__bf16)w2.y;
        bfrag[10] = (__bf16)w2.z; bfrag[11] = (__bf16)w2.w;
        bfrag[12] = (__bf16)w3.x; bfrag[13] = (__bf16)w3.y;
        bfrag[14] = (__bf16)w3.z; bfrag[15] = (__bf16)w3.w;

        acc = __builtin_amdgcn_wmma_f32_16x16x32_bf16(
            /*neg_a=*/false, afrag, /*neg_b=*/false, bfrag,
            /*c_mod=*/(short)0, acc, /*reuse_a=*/false, /*reuse_b=*/false);
    }

    const float bv = bias[n];
#pragma unroll
    for (int i = 0; i < 8; ++i) {
        float v = acc[i] + bv;
        if (relu) v = fmaxf(v, 0.0f);
        const int m = m0 + half * 8 + i;
        if (outF) outF[(size_t)m * N + n] = v;
        if (outB) outB[(size_t)m * N + n] = (__bf16)v;
    }
}

// ---------------------------------------------------------------------------
// Heads: out = [fc7 @ W_loc.T + b_loc (128x84) ; fc7 @ W_sc.T + b_sc (128x21)]
// Tiny (55 MFLOP): one block per ROI, fc7 row staged in LDS.
// ---------------------------------------------------------------------------
__global__ __launch_bounds__(128) void heads_kernel(
    const float* __restrict__ fc7,
    const float* __restrict__ Wl, const float* __restrict__ bl,
    const float* __restrict__ Ws, const float* __restrict__ bs,
    float* __restrict__ out)
{
    __shared__ float row[D2];
    const int r = blockIdx.x;
    for (int k = threadIdx.x; k < D2; k += 128)
        row[k] = fc7[(size_t)r * D2 + k];
    __syncthreads();

    const int j = threadIdx.x;
    if (j >= NLOC + NSC) return;

    const float* wr;
    float s;
    if (j < NLOC) { wr = Wl + (size_t)j * D2;          s = bl[j]; }
    else          { wr = Ws + (size_t)(j - NLOC) * D2; s = bs[j - NLOC]; }

    for (int k = 0; k < D2; ++k) s += row[k] * wr[k];

    if (j < NLOC) out[(size_t)r * NLOC + j] = s;
    else          out[(size_t)RR * NLOC + (size_t)r * NSC + (j - NLOC)] = s;
}

// ---------------------------------------------------------------------------
extern "C" void kernel_launch(void* const* d_in, const int* in_sizes, int n_in,
                              void* d_out, int out_size, void* d_ws, size_t ws_size,
                              hipStream_t stream)
{
    const float* x    = (const float*)d_in[0];
    const float* rois = (const float*)d_in[1];
    const int*   ridx = (const int*)  d_in[2];
    const float* W1   = (const float*)d_in[3];
    const float* b1   = (const float*)d_in[4];
    const float* W2   = (const float*)d_in[5];
    const float* b2   = (const float*)d_in[6];
    const float* Wl   = (const float*)d_in[7];
    const float* bl   = (const float*)d_in[8];
    const float* Wsc  = (const float*)d_in[9];
    const float* bsc  = (const float*)d_in[10];

    char* ws = (char*)d_ws;
    // Workspace layout (all 256B-aligned):
    //   featB : 128*25088 bf16 = 6,422,528 B
    //   fcB   : 128*4096  bf16 = 1,048,576 B
    //   fc7f  : 128*4096  f32  = 2,097,152 B
    __bf16* featB = (__bf16*)ws;
    __bf16* fcB   = (__bf16*)(ws + 6422528);
    float*  fc7f  = (float*) (ws + 6422528 + 1048576);
    float*  out   = (float*)d_out;

    // 1) RoI max pool -> bf16 feature matrix (128 x 25088)
    {
        const int total = RR * CC * PP * PP;
        roi_pool_kernel<<<(total + 255) / 256, 256, 0, stream>>>(x, rois, ridx, featB);
    }
    // 2) fc6: relu(feat @ W1^T + b1) -> bf16 (128 x 4096)   [411 MB weights, BW-bound]
    gemm_ws<<<D2 / 16, 256, 0, stream>>>(featB, W1, b1, fcB, nullptr, D1, D2, 1);
    // 3) fc7: relu(fc @ W2^T + b2) -> fp32 (128 x 4096)     [64 MB weights]
    gemm_ws<<<D2 / 16, 256, 0, stream>>>(fcB, W2, b2, nullptr, fc7f, D2, D2, 1);
    // 4) heads -> d_out = [cls_locs | scores]
    heads_kernel<<<RR, 128, 0, stream>>>(fc7f, Wl, bl, Wsc, bsc, out);
}
